// LinearMessagePassingLayer_89481348644992
// MI455X (gfx1250) — matrix operand order repaired and checked by reading, not compiled, not run on hardware
//
#include <hip/hip_runtime.h>
#include <hip/hip_bf16.h>

// ---------------- problem constants ----------------
constexpr long  kNodes   = 100000;
constexpr long  kEdges   = 800000;
constexpr int   kNodeDim = 128;
constexpr int   kEdgeDim = 32;
constexpr int   kMsgDim  = 128;
constexpr float kEps     = 1e-5f;

typedef __attribute__((ext_vector_type(16))) __bf16          v16bf;
typedef __attribute__((ext_vector_type(8)))  float           v8f;
typedef __attribute__((ext_vector_type(16))) unsigned short  v16u;
typedef __attribute__((ext_vector_type(8)))  unsigned short  v8u;

// ---------------- helpers ----------------
__device__ __forceinline__ unsigned short f2bfu(float f) {
  unsigned u = __builtin_bit_cast(unsigned, f);
  return (unsigned short)((u + 0x7FFFu + ((u >> 16) & 1u)) >> 16);  // RNE
}
__device__ __forceinline__ __bf16 u2bf(unsigned short s) {
  return __builtin_bit_cast(__bf16, s);
}

// A-fragment (16x32 bf16, M = lane%16): half h holds K = kb+h*8..+7 and kb+16+h*8..+7
// f32 source (converted in-register; used only for data read exactly once)
__device__ __forceinline__ v16bf a_from_f32row(const float* rowp, int kb, int h) {
  v16bf a;
  const float* p0 = rowp + kb + h * 8;
  const float* p1 = rowp + kb + 16 + h * 8;
#pragma unroll
  for (int j = 0; j < 8; ++j) a[j] = u2bf(f2bfu(p0[j]));
#pragma unroll
  for (int j = 0; j < 8; ++j) a[8 + j] = u2bf(f2bfu(p1[j]));
  return a;
}

// bf16 source (global pre-converted nodes, or LDS activation tile): two 16B loads
__device__ __forceinline__ v16bf a_from_u16row(const unsigned short* rowp, int kb, int h) {
  v8u lo = *(const v8u*)(rowp + kb + h * 8);
  v8u hi = *(const v8u*)(rowp + kb + 16 + h * 8);
  v16u raw = __builtin_shufflevector(lo, hi, 0,1,2,3,4,5,6,7,8,9,10,11,12,13,14,15);
  return __builtin_bit_cast(v16bf, raw);
}

// B-fragment from fragment-ordered packed weights: one contiguous 32B load per lane
__device__ __forceinline__ v16bf b_frag(const unsigned short* packed, int kc, int nt,
                                        int NT, int lane) {
  v16u raw = *(const v16u*)(packed + ((((long)kc * NT + nt) * 32 + lane) << 4));
  return __builtin_bit_cast(v16bf, raw);
}

// ---------------- prep: f32 -> bf16 bulk convert (nodes) ----------------
__global__ void __launch_bounds__(256)
cvt_bf16(const float* __restrict__ src, unsigned short* __restrict__ dst, long total) {
  long i = (long)blockIdx.x * 256 + threadIdx.x;
  if (i < total) dst[i] = f2bfu(src[i]);
}

// ---------------- prep: weight packing (f32 [N][K] row-major -> fragment order) ----
__global__ void __launch_bounds__(256)
pack_weight(const float* __restrict__ W, unsigned short* __restrict__ dst,
            int K, int NT, int total) {
  int idx = blockIdx.x * 256 + threadIdx.x;
  if (idx >= total) return;
  int j    = idx & 15;
  int lane = (idx >> 4) & 31;
  int tile = idx >> 9;
  int nt = tile % NT;
  int kc = tile / NT;
  int n = nt * 16 + (lane & 15);
  int k = kc * 32 + (lane >> 4) * 8 + (j & 7) + (j >> 3) * 16;
  dst[idx] = f2bfu(W[(long)n * K + k]);
}

// ---------------- edge kernel: messages + scatter-add ----------------
__global__ void __launch_bounds__(256)
edge_kernel(const unsigned short* __restrict__ nbf, const long long* __restrict__ eidx,
            const float* __restrict__ eattr, const unsigned short* __restrict__ wm,
            float* __restrict__ agg) {
  int lane = threadIdx.x & 31;
  int wave = threadIdx.x >> 5;
  int h = lane >> 4;
  int m = lane & 15;
  long tile = (long)blockIdx.x * 8 + wave;
  long e0 = tile * 16;
  long e = e0 + m;

  long long sender = eidx[e];
  const unsigned short* nrow = nbf + (long)sender * kNodeDim;
  const float* arow = eattr + e * kEdgeDim;

  // prefetch next tile's edge attributes (global_prefetch_b8)
  __builtin_prefetch(eattr + (e + 128) * kEdgeDim, 0, 1);

  v8f acc[8] = {};
#pragma unroll
  for (int kc = 0; kc < 5; ++kc) {
    // hoist all 8 B-fragments: one load clause, partial waits overlap the WMMAs
    v16bf bfr[8];
#pragma unroll
    for (int t = 0; t < 8; ++t) bfr[t] = b_frag(wm, kc, t, 8, lane);
    v16bf a = (kc < 4) ? a_from_u16row(nrow, kc * 32, h)
                       : a_from_f32row(arow, 0, h);
#pragma unroll
    for (int t = 0; t < 8; ++t)
      acc[t] = __builtin_amdgcn_wmma_f32_16x16x32_bf16(
          false, a, false, bfr[t], (short)0, acc[t], false, false);
  }

  // C layout: VGPR r, lane<16 -> row r ; lane>=16 -> row r+8 ; N = t*16 + lane%16
  long long recv[8];
#pragma unroll
  for (int r = 0; r < 8; ++r) recv[r] = eidx[kEdges + e0 + r + h * 8];

  int n0 = lane & 15;
#pragma unroll
  for (int t = 0; t < 8; ++t) {
#pragma unroll
    for (int r = 0; r < 8; ++r) {
      float v = acc[t][r];
      __hip_atomic_fetch_add(&agg[(long)recv[r] * kMsgDim + t * 16 + n0], v,
                             __ATOMIC_RELAXED, __HIP_MEMORY_SCOPE_AGENT);
    }
  }
}

// ---------------- bias + (ReLU) + LayerNorm, in C-fragment layout ----------------
__device__ __forceinline__ void bias_act_ln(v8f acc[8], const float* bias,
                                            const float* gamma, const float* beta,
                                            int lane, bool relu) {
  int n0 = lane & 15;
  float bsv[8], gv[8], bv[8];
#pragma unroll
  for (int t = 0; t < 8; ++t) {
    int n = t * 16 + n0;
    bsv[t] = bias ? bias[n] : 0.f;
    gv[t] = gamma[n];
    bv[t] = beta[n];
  }
#pragma unroll
  for (int r = 0; r < 8; ++r) {
    float s = 0.f, sq = 0.f;
#pragma unroll
    for (int t = 0; t < 8; ++t) {
      float y = acc[t][r] + bsv[t];
      if (relu) y = fmaxf(y, 0.f);
      acc[t][r] = y;
      s += y;
      sq += y * y;
    }
    // reduce across the 16 lanes of this half-wave (xor 1,2,4,8 stays in-half)
#pragma unroll
    for (int o = 1; o < 16; o <<= 1) {
      s  += __shfl_xor(s,  o, 32);
      sq += __shfl_xor(sq, o, 32);
    }
    float mean = s * (1.f / 128.f);
    float var  = sq * (1.f / 128.f) - mean * mean;
    float inv  = rsqrtf(var + kEps);
#pragma unroll
    for (int t = 0; t < 8; ++t)
      acc[t][r] = (acc[t][r] - mean) * inv * gv[t] + bv[t];
  }
}

// ---------------- fused node MLP kernel ----------------
__global__ void __launch_bounds__(256)
node_kernel(const unsigned short* __restrict__ nbf, const float* __restrict__ agg,
            const unsigned short* __restrict__ w1p, const unsigned short* __restrict__ w2p,
            const unsigned short* __restrict__ wnp,
            const float* __restrict__ b1, const float* __restrict__ g1, const float* __restrict__ bb1,
            const float* __restrict__ b2, const float* __restrict__ g2, const float* __restrict__ bb2,
            const float* __restrict__ gf, const float* __restrict__ bf_,
            float* __restrict__ out) {
  __shared__ unsigned short hls[8][16][128];  // per-wave 16x128 bf16 activation tile

  int lane = threadIdx.x & 31;
  int wave = threadIdx.x >> 5;
  int h = lane >> 4;
  int m = lane & 15;
  long base = ((long)blockIdx.x * 8 + wave) * 16;

  long rowA = base + m;
  if (rowA >= kNodes) rowA = kNodes - 1;  // clamp loads, guard stores (EXEC must stay full)
  const unsigned short* nrow = nbf + rowA * kNodeDim;
  const float* grow = agg + rowA * kMsgDim;

  // ---- stage 1: node_input [nodes | agg] @ W1^T  (K = 256) ----
  v8f acc[8] = {};
#pragma unroll
  for (int kc = 0; kc < 8; ++kc) {
    v16bf bfr[8];
#pragma unroll
    for (int t = 0; t < 8; ++t) bfr[t] = b_frag(w1p, kc, t, 8, lane);
    v16bf a = (kc < 4) ? a_from_u16row(nrow, kc * 32, h)
                       : a_from_f32row(grow, (kc - 4) * 32, h);
#pragma unroll
    for (int t = 0; t < 8; ++t)
      acc[t] = __builtin_amdgcn_wmma_f32_16x16x32_bf16(
          false, a, false, bfr[t], (short)0, acc[t], false, false);
  }
  bias_act_ln(acc, b1, g1, bb1, lane, true);

  // spill h to LDS as bf16 (wave-private slice; per-wave DS ordering suffices)
#pragma unroll
  for (int t = 0; t < 8; ++t)
#pragma unroll
    for (int r = 0; r < 8; ++r)
      hls[wave][r + h * 8][t * 16 + (lane & 15)] = f2bfu(acc[t][r]);

  // ---- stage 2: h @ W2^T  (K = 128) ----
  v8f acc2[8] = {};
  const unsigned short* hrow = &hls[wave][m][0];
#pragma unroll
  for (int kc = 0; kc < 4; ++kc) {
    v16bf bfr[8];
#pragma unroll
    for (int t = 0; t < 8; ++t) bfr[t] = b_frag(w2p, kc, t, 8, lane);
    v16bf a = a_from_u16row(hrow, kc * 32, h);
#pragma unroll
    for (int t = 0; t < 8; ++t)
      acc2[t] = __builtin_amdgcn_wmma_f32_16x16x32_bf16(
          false, a, false, bfr[t], (short)0, acc2[t], false, false);
  }
  bias_act_ln(acc2, b2, g2, bb2, lane, true);  // acc2 = node_out

  // ---- stage 3: nodes @ W_node^T + node_out, final LN ----
  v8f acc3[8] = {};
#pragma unroll
  for (int kc = 0; kc < 4; ++kc) {
    v16bf bfr[8];
#pragma unroll
    for (int t = 0; t < 8; ++t) bfr[t] = b_frag(wnp, kc, t, 8, lane);
    v16bf a = a_from_u16row(nrow, kc * 32, h);
#pragma unroll
    for (int t = 0; t < 8; ++t)
      acc3[t] = __builtin_amdgcn_wmma_f32_16x16x32_bf16(
          false, a, false, bfr[t], (short)0, acc3[t], false, false);
  }
#pragma unroll
  for (int t = 0; t < 8; ++t)
#pragma unroll
    for (int r = 0; r < 8; ++r) acc3[t][r] += acc2[t][r];
  bias_act_ln(acc3, nullptr, gf, bf_, lane, false);

  // store (predicated for the grid tail)
#pragma unroll
  for (int r = 0; r < 8; ++r) {
    long row = base + r + h * 8;
    if (row < kNodes) {
#pragma unroll
      for (int t = 0; t < 8; ++t)
        out[row * kNodeDim + t * 16 + m] = acc3[t][r];
    }
  }
}

// ---------------- launcher ----------------
extern "C" void kernel_launch(void* const* d_in, const int* in_sizes, int n_in,
                              void* d_out, int out_size, void* d_ws, size_t ws_size,
                              hipStream_t stream) {
  // input order per setup_inputs():
  const float*     nodes   = (const float*)d_in[0];
  const long long* eidx    = (const long long*)d_in[1];   // int64 [2, E]
  const float*     eattr   = (const float*)d_in[2];
  const float*     Wmsg    = (const float*)d_in[3];       // [128, 160]
  const float*     Wnode   = (const float*)d_in[4];       // [128, 128]
  const float*     W1      = (const float*)d_in[5];       // [128, 256]
  const float*     b1      = (const float*)d_in[6];
  const float*     g1      = (const float*)d_in[7];
  const float*     bb1     = (const float*)d_in[8];
  const float*     W2      = (const float*)d_in[9];       // [128, 128]
  const float*     b2      = (const float*)d_in[10];
  const float*     g2      = (const float*)d_in[11];
  const float*     bb2     = (const float*)d_in[12];
  const float*     gf      = (const float*)d_in[13];
  const float*     bf_     = (const float*)d_in[14];
  float*           out     = (float*)d_out;

  // workspace layout
  const long aggElems  = kNodes * (long)kMsgDim;   // 12.8M f32
  const long nodeElems = kNodes * (long)kNodeDim;  // 12.8M bf16
  char* ws = (char*)d_ws;
  float*          agg  = (float*)ws;               ws += aggElems * sizeof(float);
  unsigned short* nbf  = (unsigned short*)ws;      ws += nodeElems * 2;
  unsigned short* wmP  = (unsigned short*)ws;      ws += 5 * 8 * 512 * 2;  // K=160
  unsigned short* w1P  = (unsigned short*)ws;      ws += 8 * 8 * 512 * 2;  // K=256
  unsigned short* w2P  = (unsigned short*)ws;      ws += 4 * 8 * 512 * 2;  // K=128
  unsigned short* wnP  = (unsigned short*)ws;      ws += 4 * 8 * 512 * 2;  // K=128

  // zero aggregation buffer every call (atomics accumulate; graph replays this)
  hipMemsetAsync(agg, 0, aggElems * sizeof(float), stream);

  // pre-convert nodes to bf16 (gathered ~8x by edges + read 2x by node MLP)
  cvt_bf16<<<(int)((nodeElems + 255) / 256), 256, 0, stream>>>(nodes, nbf, nodeElems);

  // re-pack weights to fragment-ordered bf16 (deterministic, tiny)
  {
    int t;
    t = 5 * 8 * 512; pack_weight<<<(t + 255) / 256, 256, 0, stream>>>(Wmsg,  wmP, 160, 8, t);
    t = 8 * 8 * 512; pack_weight<<<(t + 255) / 256, 256, 0, stream>>>(W1,    w1P, 256, 8, t);
    t = 4 * 8 * 512; pack_weight<<<(t + 255) / 256, 256, 0, stream>>>(W2,    w2P, 128, 8, t);
    t = 4 * 8 * 512; pack_weight<<<(t + 255) / 256, 256, 0, stream>>>(Wnode, wnP, 128, 8, t);
  }

  // edge messages + scatter: 800000 edges, 128 per block (8 waves x 16 edges)
  edge_kernel<<<(int)(kEdges / 128), 256, 0, stream>>>(nbf, eidx, eattr, wmP, agg);

  // fused node MLP: 128 nodes per block
  int nodeBlocks = (int)((kNodes + 127) / 128);
  node_kernel<<<nodeBlocks, 256, 0, stream>>>(nbf, agg, w1P, w2P, wnP,
                                              b1, g1, bb1, b2, g2, bb2, gf, bf_, out);
}
